// InternImageLayer_84104049590498
// MI455X (gfx1250) — matrix operand order repaired and checked by reading, not compiled, not run on hardware
//
#include <hip/hip_runtime.h>
#include <hip/hip_bf16.h>

// ---------------------------------------------------------------------------
// InternImage / DCNv3 block for gfx1250 (MI455X).
// All channel-mixing GEMMs run through v_wmma_f32_16x16x32_f16 (wave32 WMMA),
// f16 operands with fp32 accumulation, with N-register-blocking (NT tiles per
// wave share one A fragment per K-step). Pointwise stages are wave-per-token.
// ---------------------------------------------------------------------------

#define NB   4
#define HH   128
#define WW   128
#define CC   128
#define MTOK (NB * HH * WW)   // 65536 tokens
#define GG   8
#define GC   16
#define KK   9

typedef _Float16 h4  __attribute__((ext_vector_type(4)));
typedef _Float16 h8  __attribute__((ext_vector_type(8)));
typedef _Float16 h16 __attribute__((ext_vector_type(16)));
typedef float    f8  __attribute__((ext_vector_type(8)));

// ---------------------------------------------------------------------------
// Wave (32-lane) sum reduction
// ---------------------------------------------------------------------------
__device__ __forceinline__ float waveReduceSum(float v) {
#pragma unroll
    for (int off = 16; off > 0; off >>= 1)
        v += __shfl_xor(v, off, 32);
    return v;
}

__device__ __forceinline__ float gelu_exact(float v) {
    return 0.5f * v * (1.0f + erff(v * 0.70710678118654752f));
}

// ---------------------------------------------------------------------------
// Weight prep: fp32 [K x N] row-major  ->  f16 Bt [Npad x K], zero padded rows
// ---------------------------------------------------------------------------
__global__ __launch_bounds__(256)
void prep_weight_kernel(const float* __restrict__ W, _Float16* __restrict__ Bt,
                        int K, int N, int Npad) {
    int idx = blockIdx.x * blockDim.x + threadIdx.x;
    if (idx >= Npad * K) return;
    int nn = idx / K;
    int k  = idx - nn * K;
    float v = (nn < N) ? W[(size_t)k * N + nn] : 0.0f;
    Bt[idx] = (_Float16)v;
}

// ---------------------------------------------------------------------------
// LayerNorm over C=128, one wave per token (4 channels per lane).
// Optional fp32 and f16 outputs.
// ---------------------------------------------------------------------------
__global__ __launch_bounds__(256)
void ln_kernel(const float* __restrict__ in, const float* __restrict__ g,
               const float* __restrict__ b, float* __restrict__ outF,
               _Float16* __restrict__ outH, int Mtok) {
    int wid = blockIdx.x * (blockDim.x >> 5) + (threadIdx.x >> 5);
    if (wid >= Mtok) return;
    int lane = threadIdx.x & 31;
    const float* row = in + (size_t)wid * CC;
    float4 v = *(const float4*)(row + lane * 4);
    float s  = v.x + v.y + v.z + v.w;
    float s2 = v.x * v.x + v.y * v.y + v.z * v.z + v.w * v.w;
    s  = waveReduceSum(s);
    s2 = waveReduceSum(s2);
    float mean = s * (1.0f / 128.0f);
    float var  = s2 * (1.0f / 128.0f) - mean * mean;
    float rstd = rsqrtf(var + 1e-5f);
    float4 gv = *(const float4*)(g + lane * 4);
    float4 bv = *(const float4*)(b + lane * 4);
    float o0 = (v.x - mean) * rstd * gv.x + bv.x;
    float o1 = (v.y - mean) * rstd * gv.y + bv.y;
    float o2 = (v.z - mean) * rstd * gv.z + bv.z;
    float o3 = (v.w - mean) * rstd * gv.w + bv.w;
    if (outF) {
        float4 o = make_float4(o0, o1, o2, o3);
        *(float4*)(outF + (size_t)wid * CC + lane * 4) = o;
    }
    if (outH) {
        h4 o = { (_Float16)o0, (_Float16)o1, (_Float16)o2, (_Float16)o3 };
        *(h4*)(outH + (size_t)wid * CC + lane * 4) = o;
    }
}

// ---------------------------------------------------------------------------
// Depthwise 3x3 conv (SAME, zero pad) + LayerNorm + exact GELU -> f16
// One wave per token, 4 channels per lane.
// ---------------------------------------------------------------------------
__global__ __launch_bounds__(256)
void dwconv_ln_gelu_kernel(const float* __restrict__ xln,
                           const float* __restrict__ kwt,   // (3*3, C)
                           const float* __restrict__ kb,
                           const float* __restrict__ lng,
                           const float* __restrict__ lnb,
                           _Float16* __restrict__ x1) {
    int wid = blockIdx.x * (blockDim.x >> 5) + (threadIdx.x >> 5);
    if (wid >= MTOK) return;
    int lane = threadIdx.x & 31;
    int n  = wid >> 14;
    int hw = wid & 16383;
    int h  = hw >> 7;
    int w  = hw & 127;
    int c  = lane * 4;

    float4 bb = *(const float4*)(kb + c);
    float a0 = bb.x, a1 = bb.y, a2 = bb.z, a3 = bb.w;
#pragma unroll
    for (int dy = -1; dy <= 1; dy++) {
#pragma unroll
        for (int dx = -1; dx <= 1; dx++) {
            int hh = h + dy, wx = w + dx;
            if ((unsigned)hh < (unsigned)HH && (unsigned)wx < (unsigned)WW) {
                const float4 xv = *(const float4*)(
                    xln + (((size_t)n * (HH * WW) + hh * WW + wx) * CC) + c);
                const float4 kv = *(const float4*)(
                    kwt + ((dy + 1) * 3 + (dx + 1)) * CC + c);
                a0 += xv.x * kv.x; a1 += xv.y * kv.y;
                a2 += xv.z * kv.z; a3 += xv.w * kv.w;
            }
        }
    }
    // LayerNorm across the 128 conv outputs
    float s  = a0 + a1 + a2 + a3;
    float s2 = a0 * a0 + a1 * a1 + a2 * a2 + a3 * a3;
    s  = waveReduceSum(s);
    s2 = waveReduceSum(s2);
    float mean = s * (1.0f / 128.0f);
    float var  = s2 * (1.0f / 128.0f) - mean * mean;
    float rstd = rsqrtf(var + 1e-5f);
    float4 gv = *(const float4*)(lng + c);
    float4 bv = *(const float4*)(lnb + c);
    float o0 = gelu_exact((a0 - mean) * rstd * gv.x + bv.x);
    float o1 = gelu_exact((a1 - mean) * rstd * gv.y + bv.y);
    float o2 = gelu_exact((a2 - mean) * rstd * gv.z + bv.z);
    float o3 = gelu_exact((a3 - mean) * rstd * gv.w + bv.w);
    h4 o = { (_Float16)o0, (_Float16)o1, (_Float16)o2, (_Float16)o3 };
    *(h4*)(x1 + (size_t)wid * CC + c) = o;
}

// ---------------------------------------------------------------------------
// Templated WMMA GEMM: out[M x N] = A[M x K] * Bt[N x K]^T (+ epilogue)
// One wave computes NT consecutive 16x16 tiles along N, sharing a single
// A fragment per 32-deep K-step (register blocking). K is compile-time so
// the K-loop fully unrolls into load clauses + back-to-back WMMAs.
// MODE 0: + bias -> fp32 | MODE 1: + bias, GELU -> f16 | MODE 2: + bias + resid -> fp32
// ---------------------------------------------------------------------------
template <int MODE, int NT, int KV>
__global__ __launch_bounds__(256)
void wmma_gemm_kernel(const _Float16* __restrict__ A,
                      const _Float16* __restrict__ Bt,
                      const float* __restrict__ bias,
                      const float* __restrict__ resid,
                      float* __restrict__ outF,
                      _Float16* __restrict__ outH,
                      int Nreal) {
    const int lane  = threadIdx.x & 31;
    const int wave  = threadIdx.x >> 5;
    const int tileM = blockIdx.y * 8 + wave;
    const int nBase = blockIdx.x * NT;          // first 16-wide N tile

    // A fragment addressing (ISA 16-bit A 16x32 layout):
    //   lane<16 : halves 0-7 = K 0..7,  halves 8-15 = K 16..23 (row M = lane)
    //   lane>=16: halves 0-7 = K 8..15, halves 8-15 = K 24..31 (row M = lane-16)
    const int mRow = tileM * 16 + (lane & 15);
    const int aKlo = (lane & 16) ? 8 : 0;
    // B fragment addressing (ISA B K x 16 layout):
    //   lane<16 : K 0..7,16..23 for column N = lane ; lane>=16: K 8..15,24..31
    const int bKlo = (lane & 16) ? 8 : 0;

    const _Float16* aPtr = A  + (size_t)mRow * KV + aKlo;
    const _Float16* bPtr = Bt + ((size_t)(nBase * 16 + (lane & 15))) * KV + bKlo;

    f8 c[NT];
#pragma unroll
    for (int t = 0; t < NT; t++) c[t] = {};

#pragma unroll
    for (int kk = 0; kk < KV; kk += 32) {
        h8  a0 = *(const h8*)(aPtr + kk);
        h8  a1 = *(const h8*)(aPtr + kk + 16);
        h16 af;
#pragma unroll
        for (int i = 0; i < 8; i++) { af[i] = a0[i]; af[i + 8] = a1[i]; }
#pragma unroll
        for (int t = 0; t < NT; t++) {
            h8 b0 = *(const h8*)(bPtr + (size_t)t * 16 * KV + kk);
            h8 b1 = *(const h8*)(bPtr + (size_t)t * 16 * KV + kk + 16);
            h16 bf;
#pragma unroll
            for (int i = 0; i < 8; i++) { bf[i] = b0[i]; bf[i + 8] = b1[i]; }
            c[t] = __builtin_amdgcn_wmma_f32_16x16x32_f16(
                       false, af, false, bf, (short)0, c[t], false, false);
        }
    }

    // C/D layout: VGPR r -> row (lane<16 ? r : r+8), col = lane&15
    const int rowBase = tileM * 16 + ((lane & 16) ? 8 : 0);
#pragma unroll
    for (int t = 0; t < NT; t++) {
        const int col = (nBase + t) * 16 + (lane & 15);
        if (col < Nreal) {
            const float bval = bias[col];
#pragma unroll
            for (int r = 0; r < 8; r++) {
                int row = rowBase + r;
                float v = c[t][r] + bval;
                if (MODE == 1) {
                    v = gelu_exact(v);
                    outH[(size_t)row * Nreal + col] = (_Float16)v;
                } else if (MODE == 2) {
                    v += resid[(size_t)row * Nreal + col];
                    outF[(size_t)row * Nreal + col] = v;
                } else {
                    outF[(size_t)row * Nreal + col] = v;
                }
            }
        }
    }
}

// ---------------------------------------------------------------------------
// Softmax over K=9 per (token, group), in place on [M x 72]
// ---------------------------------------------------------------------------
__global__ __launch_bounds__(256)
void softmax9_kernel(float* __restrict__ m) {
    int t = blockIdx.x * blockDim.x + threadIdx.x;
    if (t >= MTOK * GG) return;
    float* p = m + (size_t)(t >> 3) * (GG * KK) + (t & 7) * KK;
    float mx = -3.0e38f;
#pragma unroll
    for (int k = 0; k < KK; k++) mx = fmaxf(mx, p[k]);
    float e[KK], s = 0.0f;
#pragma unroll
    for (int k = 0; k < KK; k++) { e[k] = expf(p[k] - mx); s += e[k]; }
    float inv = 1.0f / s;
#pragma unroll
    for (int k = 0; k < KK; k++) p[k] = e[k] * inv;
}

// ---------------------------------------------------------------------------
// DCNv3 core: bilinear gather with offsets + mask, one thread per (token, g).
// Padded-image math folds to: px = w + 1 + kx + off_x ; valid iff 1<=ix<=W.
// ---------------------------------------------------------------------------
__global__ __launch_bounds__(256)
void dcn_core_kernel(const float* __restrict__ xp,
                     const float* __restrict__ off,
                     const float* __restrict__ mask,
                     _Float16* __restrict__ y) {
    int t = blockIdx.x * blockDim.x + threadIdx.x;
    if (t >= MTOK * GG) return;
    int tok = t >> 3;
    int g   = t & 7;
    int n   = tok >> 14;
    int h   = (tok >> 7) & 127;
    int w   = tok & 127;

    float acc[GC];
#pragma unroll
    for (int j = 0; j < GC; j++) acc[j] = 0.0f;

    const float* offp = off  + (size_t)tok * (GG * KK * 2) + g * (KK * 2);
    const float* mp   = mask + (size_t)tok * (GG * KK)     + g * KK;

#pragma unroll
    for (int k = 0; k < KK; k++) {
        float ox = offp[k * 2 + 0];
        float oy = offp[k * 2 + 1];
        float kx = (float)(k / 3) - 1.0f;   // x-major kernel grid
        float ky = (float)(k % 3) - 1.0f;
        float px = (float)w + 1.0f + kx + ox;
        float py = (float)h + 1.0f + ky + oy;
        float x0 = floorf(px), y0 = floorf(py);
        float tx = px - x0, ty = py - y0;
        int x0i = (int)x0, y0i = (int)y0;
        float mk = mp[k];

        float cw[4] = { (1.0f - ty) * (1.0f - tx) * mk,
                        (1.0f - ty) * tx * mk,
                        ty * (1.0f - tx) * mk,
                        ty * tx * mk };
        int cy[4] = { y0i, y0i, y0i + 1, y0i + 1 };
        int cx[4] = { x0i, x0i + 1, x0i, x0i + 1 };
#pragma unroll
        for (int cn = 0; cn < 4; cn++) {
            int iy = cy[cn], ix = cx[cn];
            if (iy >= 1 && iy <= HH && ix >= 1 && ix <= WW) {
                const float* src = xp +
                    (((size_t)n * (HH * WW) + (size_t)(iy - 1) * WW + (ix - 1)) * CC)
                    + g * GC;
                float wt = cw[cn];
#pragma unroll
                for (int q = 0; q < 4; q++) {
                    float4 vv = *(const float4*)(src + q * 4);
                    acc[q * 4 + 0] += wt * vv.x;
                    acc[q * 4 + 1] += wt * vv.y;
                    acc[q * 4 + 2] += wt * vv.z;
                    acc[q * 4 + 3] += wt * vv.w;
                }
            }
        }
    }

    _Float16* dst = y + (size_t)tok * CC + g * GC;
#pragma unroll
    for (int q = 0; q < 4; q++) {
        h4 o = { (_Float16)acc[q * 4 + 0], (_Float16)acc[q * 4 + 1],
                 (_Float16)acc[q * 4 + 2], (_Float16)acc[q * 4 + 3] };
        *(h4*)(dst + q * 4) = o;
    }
}

// ---------------------------------------------------------------------------
// Launch
// ---------------------------------------------------------------------------
extern "C" void kernel_launch(void* const* d_in, const int* in_sizes, int n_in,
                              void* d_out, int out_size, void* d_ws, size_t ws_size,
                              hipStream_t stream) {
    const float* x      = (const float*)d_in[0];
    const float* ln1_g  = (const float*)d_in[1];
    const float* ln1_b  = (const float*)d_in[2];
    const float* in_w   = (const float*)d_in[3];
    const float* in_b   = (const float*)d_in[4];
    const float* dw_k   = (const float*)d_in[5];
    const float* dw_b   = (const float*)d_in[6];
    const float* dwln_g = (const float*)d_in[7];
    const float* dwln_b = (const float*)d_in[8];
    const float* off_w  = (const float*)d_in[9];
    const float* off_b  = (const float*)d_in[10];
    const float* mask_w = (const float*)d_in[11];
    const float* mask_b = (const float*)d_in[12];
    const float* out_w  = (const float*)d_in[13];
    const float* out_b  = (const float*)d_in[14];
    const float* ln2_g  = (const float*)d_in[15];
    const float* ln2_b  = (const float*)d_in[16];
    const float* fc1_w  = (const float*)d_in[17];
    const float* fc1_b  = (const float*)d_in[18];
    const float* fc2_w  = (const float*)d_in[19];
    const float* fc2_b  = (const float*)d_in[20];
    float* out = (float*)d_out;

    char* ws = (char*)d_ws;
    size_t wo = 0;
    auto alloc = [&](size_t bytes) -> void* {
        void* p = ws + wo;
        wo = (wo + bytes + 255) & ~(size_t)255;
        return p;
    };

    const size_t M = MTOK;
    float*    xln_f32 = (float*)   alloc(M * CC * 4);
    _Float16* xln_f16 = (_Float16*)alloc(M * CC * 2);
    float*    xp      = (float*)   alloc(M * CC * 4);
    _Float16* x1_f16  = (_Float16*)alloc(M * CC * 2);
    float*    off_buf = (float*)   alloc(M * 144 * 4);
    float*    mask_buf= (float*)   alloc(M * 72 * 4);
    _Float16* y_f16   = (_Float16*)alloc(M * CC * 2);
    float*    x2      = (float*)   alloc(M * CC * 4);
    _Float16* h_f16   = (_Float16*)alloc(M * CC * 2);
    _Float16* u_f16   = (_Float16*)alloc(M * 512 * 2);
    _Float16* in_wT   = (_Float16*)alloc(128 * 128 * 2);
    _Float16* off_wT  = (_Float16*)alloc(144 * 128 * 2);
    _Float16* mask_wT = (_Float16*)alloc(80 * 128 * 2);   // padded 72 -> 80
    _Float16* out_wT  = (_Float16*)alloc(128 * 128 * 2);
    _Float16* fc1_wT  = (_Float16*)alloc(512 * 128 * 2);
    _Float16* fc2_wT  = (_Float16*)alloc(128 * 512 * 2);
    (void)ws_size; (void)in_sizes; (void)n_in; (void)out_size;

    // --- weight prep (transpose + cast to f16) ---
    auto prep = [&](const float* W, _Float16* Bt, int K, int N, int Npad) {
        int total = Npad * K;
        prep_weight_kernel<<<(total + 255) / 256, 256, 0, stream>>>(W, Bt, K, N, Npad);
    };
    prep(in_w,   in_wT,   128, 128, 128);
    prep(off_w,  off_wT,  128, 144, 144);
    prep(mask_w, mask_wT, 128, 72,  80);
    prep(out_w,  out_wT,  128, 128, 128);
    prep(fc1_w,  fc1_wT,  128, 512, 512);
    prep(fc2_w,  fc2_wT,  512, 128, 128);

    const int lnGrid  = (int)(M / 8);     // 8 waves per 256-thread block
    const int mBlocks = (int)(M / 128);   // 8 M-tiles per block

    // --- LN1: x -> xln (fp32 + f16) ---
    ln_kernel<<<lnGrid, 256, 0, stream>>>(x, ln1_g, ln1_b, xln_f32, xln_f16, (int)M);

    // --- in_proj: xp = xln @ in_w + in_b (fp32, for sampling) ---
    wmma_gemm_kernel<0, 4, 128><<<dim3(2, mBlocks), 256, 0, stream>>>(
        xln_f16, in_wT, in_b, nullptr, xp, nullptr, 128);

    // --- depthwise conv + LN + GELU -> x1 (f16) ---
    dwconv_ln_gelu_kernel<<<lnGrid, 256, 0, stream>>>(
        xln_f32, dw_k, dw_b, dwln_g, dwln_b, x1_f16);

    // --- offset = x1 @ off_w + off_b (fp32), 9 tiles = 3 blocks x NT=3 ---
    wmma_gemm_kernel<0, 3, 128><<<dim3(3, mBlocks), 256, 0, stream>>>(
        x1_f16, off_wT, off_b, nullptr, off_buf, nullptr, 144);

    // --- mask logits = x1 @ mask_w + mask_b (fp32, N padded 72->80, NT=5) ---
    wmma_gemm_kernel<0, 5, 128><<<dim3(1, mBlocks), 256, 0, stream>>>(
        x1_f16, mask_wT, mask_b, nullptr, mask_buf, nullptr, 72);

    // --- softmax over K=9 (in place) ---
    softmax9_kernel<<<(int)(M * GG / 256), 256, 0, stream>>>(mask_buf);

    // --- DCN bilinear gather -> y (f16) ---
    dcn_core_kernel<<<(int)(M * GG / 256), 256, 0, stream>>>(
        xp, off_buf, mask_buf, y_f16);

    // --- out_proj + residual: x2 = y @ out_w + out_b + x ---
    wmma_gemm_kernel<2, 4, 128><<<dim3(2, mBlocks), 256, 0, stream>>>(
        y_f16, out_wT, out_b, x, x2, nullptr, 128);

    // --- LN2: x2 -> h (f16) ---
    ln_kernel<<<lnGrid, 256, 0, stream>>>(x2, ln2_g, ln2_b, nullptr, h_f16, (int)M);

    // --- fc1 + GELU: u = gelu(h @ fc1_w + fc1_b) (f16) ---
    wmma_gemm_kernel<1, 4, 128><<<dim3(8, mBlocks), 256, 0, stream>>>(
        h_f16, fc1_wT, fc1_b, nullptr, nullptr, u_f16, 512);

    // --- fc2 + residual: out = u @ fc2_w + fc2_b + x2 ---
    wmma_gemm_kernel<2, 4, 512><<<dim3(2, mBlocks), 256, 0, stream>>>(
        u_f16, fc2_wT, fc2_b, x2, out, nullptr, 128);
}
